// PostProcessWrapper_50680614093382
// MI455X (gfx1250) — compile-verified, matrix operand order checked by reading, not run on hardware
//
#include <hip/hip_runtime.h>
#include <hip/hip_bf16.h>
#include <stdint.h>

#define DEVINL __device__ __forceinline__

// ---------------- problem constants ----------------
#define BATCH     32
#define NBOX      8400
#define NCLS      80
#define NCF       672000      // NBOX*NCLS scores per batch
#define F4PB      168000      // NCF/4 float4 per batch
#define CHUNKS    657         // ceil(F4PB / 256)
#define HBLK      32          // streaming workgroups per batch
#define KTOP      1024
#define MAXDET    300
#define SCORE_TH  0.001f
#define IOU_TH    0.7f

// ---------------- workspace layout (32-bit word offsets) ----------------
#define WS_HIST   0u                      // BATCH*2048 histogram bins
#define WS_PREFIX (WS_HIST + 65536u)      // BATCH  radix prefix / threshold key
#define WS_REMAIN (WS_PREFIX + 32u)       // BATCH  remaining k in current bin
#define WS_CNT    (WS_REMAIN + 32u)       // BATCH  candidate count
#define WS_CKEY   (WS_CNT + 32u)          // BATCH*2048 candidate score bits
#define WS_CIDX   (WS_CKEY + 65536u)      // BATCH*2048 candidate flat index
#define WS_TSCORE (WS_CIDX + 65536u)      // BATCH*1024 sorted scores (f32)
#define WS_TCLS   (WS_TSCORE + 32768u)    // BATCH*1024 class ids
#define WS_TBOX   (WS_TCLS + 32768u)      // BATCH*1024*4 original boxes (16B aligned)
#define WS_TOFF   (WS_TBOX + 131072u)     // BATCH*1024*4 class-offset boxes

typedef unsigned uint32x4 __attribute__((ext_vector_type(4)));
typedef unsigned uint32x8 __attribute__((ext_vector_type(8)));
typedef _Float16 v16h __attribute__((ext_vector_type(16)));
typedef float    v8f  __attribute__((ext_vector_type(8)));

// ---------------- CDNA5 async global->LDS helpers (gfx1250) ----------------
DEVINL void async_load_f4(void* ldsDst, const void* gSrc) {
  unsigned l = (unsigned)(unsigned long long)ldsDst;        // low 32b = LDS byte addr
  unsigned long long g = (unsigned long long)gSrc;
  asm volatile("global_load_async_to_lds_b128 %0, %1, off"
               :: "v"(l), "v"(g) : "memory");
}
DEVINL void wait_async_le1() { asm volatile("s_wait_asynccnt 0x1" ::: "memory"); }
DEVINL void wait_async_le0() { asm volatile("s_wait_asynccnt 0x0" ::: "memory"); }

DEVINL unsigned score_key(float s) {
  return (s > SCORE_TH) ? __float_as_uint(s) : 0u;          // monotonic for s>0
}

// ---------------- kernels ----------------
__global__ void nms_init(unsigned* __restrict__ ws) {
  int i = blockIdx.x * 256 + threadIdx.x;
  if (i < 65536) ws[WS_HIST + i] = 0u;
  if (i < BATCH) {
    ws[WS_PREFIX + i] = 0u;
    ws[WS_REMAIN + i] = (unsigned)KTOP;
    ws[WS_CNT + i]    = 0u;
  }
}

// Streaming histogram pass (2 radix rounds). Round 0 streams 86MB from HBM
// (~3.7us floor at 23.3TB/s); round 1 + collect are L2-resident (86MB << 192MB
// L2). Double-buffered async global->LDS pipeline hides latency without
// VGPR-held prefetch.
__global__ void nms_hist(const float* __restrict__ scores,
                         unsigned* __restrict__ ws, int round) {
  __shared__ unsigned lhist[2048];
  __shared__ float4 buf[2][256];
  const int tid = threadIdx.x;
  const int b   = blockIdx.y;
  for (int i = tid; i < 2048; i += 256) lhist[i] = 0u;
  __syncthreads();
  const unsigned prefix = ws[WS_PREFIX + b];
  const float4* sp = reinterpret_cast<const float4*>(scores) + (size_t)b * F4PB;

  int c = blockIdx.x;                       // chunk index, strided by HBLK
  { // prologue: issue first tile
    int g = c * 256 + tid;
    if (g < F4PB) async_load_f4(&buf[0][tid], sp + g);
  }
  int nb = 0;
  for (; c < CHUNKS; c += HBLK) {
    const int  cn   = c + HBLK;
    const bool more = (cn < CHUNKS);        // uniform per block
    if (more) {
      int g2 = cn * 256 + tid;
      if (g2 < F4PB) async_load_f4(&buf[nb ^ 1][tid], sp + g2);
    }
    if (more) wait_async_le1(); else wait_async_le0();
    const int g = c * 256 + tid;
    if (g < F4PB) {
      float4 v = buf[nb][tid];              // each thread reads only its own slot
      const float sv[4] = {v.x, v.y, v.z, v.w};
      #pragma unroll
      for (int k = 0; k < 4; ++k) {
        unsigned key = score_key(sv[k]);
        unsigned bin; bool ok;
        if (round == 0) { ok = true;                            bin = key >> 20; }
        else            { ok = ((key >> 20) == (prefix >> 20)); bin = (key >> 9) & 2047u; }
        if (ok) atomicAdd(&lhist[bin], 1u);
      }
    }
    nb ^= 1;
  }
  __syncthreads();
  for (int i = tid; i < 2048; i += 256) {
    unsigned h = lhist[i];
    if (h) atomicAdd(&ws[WS_HIST + b * 2048 + i], h);
  }
}

// Pick the radix bin containing the k-th value, update prefix, re-zero hist.
__global__ void nms_scan_bins(unsigned* __restrict__ ws, int round) {
  const int b = blockIdx.x, lane = threadIdx.x;   // 32 threads
  unsigned* h = ws + WS_HIST + b * 2048;
  if (lane == 0) {
    const int shift = (round == 0) ? 20 : 9;
    unsigned rem = ws[WS_REMAIN + b];
    unsigned cum = 0; int chosen = 0; unsigned remNew = rem;
    for (int i = 2047; i >= 0; --i) {
      unsigned cnt = h[i];
      if (cum + cnt >= rem) { chosen = i; remNew = rem - cum; break; }
      cum += cnt;
    }
    ws[WS_PREFIX + b] |= ((unsigned)chosen) << shift;
    ws[WS_REMAIN + b]  = remNew;
  }
  __syncthreads();
  for (int i = lane; i < 2048; i += 32) h[i] = 0u;
}

// Collect every entry with key >= round-1 threshold (<= ~1050 expected, cap
// 2048); the bitonic sort resolves exact top-1024 incl. tie-break by index.
__global__ void nms_collect(const float* __restrict__ scores,
                            unsigned* __restrict__ ws) {
  const int b = blockIdx.y, tid = threadIdx.x;
  const unsigned T = ws[WS_PREFIX + b];
  const float4* sp = reinterpret_cast<const float4*>(scores) + (size_t)b * F4PB;
  unsigned* ckey = ws + WS_CKEY + b * 2048;
  unsigned* cidx = ws + WS_CIDX + b * 2048;
  for (int i4 = blockIdx.x * 256 + tid; i4 < F4PB; i4 += HBLK * 256) {
    float4 v = sp[i4];
    const float sv[4] = {v.x, v.y, v.z, v.w};
    #pragma unroll
    for (int k = 0; k < 4; ++k) {
      unsigned key = score_key(sv[k]);
      if (key >= T && key != 0u) {
        unsigned p = atomicAdd(&ws[WS_CNT + b], 1u);
        if (p < 2048u) { ckey[p] = key; cidx[p] = (unsigned)(i4 * 4 + k); }
      }
    }
  }
}

// Bitonic sort 2048 packed (score desc, index asc) keys; emit sorted top-1024
// with gathered boxes, class offsets and scores.
__global__ void nms_sort(const float* __restrict__ boxes, unsigned* __restrict__ ws) {
  __shared__ unsigned long long sk[2048];
  const int b = blockIdx.x, tid = threadIdx.x;        // 1024 threads
  const unsigned n = ws[WS_CNT + b] < 2048u ? ws[WS_CNT + b] : 2048u;
  const unsigned* ckey = ws + WS_CKEY + b * 2048;
  const unsigned* cidx = ws + WS_CIDX + b * 2048;
  {
    unsigned long long v = 0ull, w = 0ull;
    if ((unsigned)tid < n)
      v = ((unsigned long long)ckey[tid] << 32) | (unsigned)~cidx[tid];
    if ((unsigned)(tid + 1024) < n)
      w = ((unsigned long long)ckey[1024 + tid] << 32) | (unsigned)~cidx[1024 + tid];
    sk[tid] = v;
    sk[1024 + tid] = w;
  }
  __syncthreads();
  for (int kk = 2; kk <= 2048; kk <<= 1) {
    for (int jj = kk >> 1; jj > 0; jj >>= 1) {
      for (int t = tid; t < 2048; t += 1024) {
        int ixj = t ^ jj;
        if (ixj > t) {
          unsigned long long a = sk[t], c = sk[ixj];
          bool descSeg = ((t & kk) == 0);
          if (descSeg ? (a < c) : (a > c)) { sk[t] = c; sk[ixj] = a; }
        }
      }
      __syncthreads();
    }
  }
  { // emit
    unsigned long long v = sk[tid];
    unsigned key = (unsigned)(v >> 32);
    unsigned idx = ~(unsigned)v;
    if (key == 0u) idx = 0u;                          // degenerate guard
    float sc = __uint_as_float(key);
    int bi = (int)(idx / NCLS), cl = (int)(idx % NCLS);
    float4 bx = reinterpret_cast<const float4*>(boxes)[(size_t)b * NBOX + bi];
    float off = (float)cl * 4096.0f;
    float4 ob; ob.x = bx.x + off; ob.y = bx.y + off; ob.z = bx.z + off; ob.w = bx.w + off;
    float* wf = (float*)ws;
    wf[WS_TSCORE + b * 1024 + tid] = sc;
    ((int*)ws)[WS_TCLS + b * 1024 + tid] = cl;
    reinterpret_cast<float4*>(wf + WS_TBOX)[b * 1024 + tid] = bx;
    reinterpret_cast<float4*>(wf + WS_TOFF)[b * 1024 + tid] = ob;
  }
}

// Fused tail: one workgroup per batch.
//  (a) TDM tensor_load_to_lds pulls the 16KB offset-box tile into LDS.
//  (b) 32 waves build the 1024x1024 IoU>0.7 bitmask in LDS (128KB — only
//      possible thanks to CDNA5's 320KB/WGP LDS) with wave32 ballots.
//  (c) wave 0 runs the order-dependent greedy scan entirely in LDS with a
//      v_readlane critical chain and next-row prefetch.
//  (d) stable compaction using a WMMA prefix-scan (lower-triangular ones
//      matrix, v_wmma_f32_16x16x32_f16) over the 32 per-wave kept counts;
//      kept entries first (score order), suppressed in index order ==
//      reference top_k over zeroed scores. Writes the final outputs.
__global__ void nms_tail(unsigned* __restrict__ ws,
                         float* __restrict__ obox, float* __restrict__ oscore,
                         int* __restrict__ olab, int* __restrict__ onval) {
  __shared__ float4   soff[1024];        // 16KB
  __shared__ float    sarea[1024];       // 4KB
  __shared__ unsigned sup[1024 * 32];    // 128KB suppression bitmask
  __shared__ unsigned keepw_s[32];
  __shared__ int   wcnt[32];
  __shared__ float sincl[32];            // inclusive prefix sums from WMMA
  const int b = blockIdx.x, tid = threadIdx.x;       // 1024 threads, 32 waves
  const int lane = tid & 31, wv = tid >> 5;
  float* wf = (float*)ws;

  // (a) TDM: 2D descriptor, 8-byte elements, tile 2048x1 (= 1024 float4 boxes)
  if (tid == 0) {
    unsigned long long ga = (unsigned long long)(wf + WS_TOFF + (size_t)b * 4096);
    uint32x4 g0;
    g0.x = 1u;                                        // count=1, load, no gather
    g0.y = (unsigned)(unsigned long long)(void*)&soff[0];   // lds_addr
    g0.z = (unsigned)ga;                              // global_addr[31:0]
    g0.w = (unsigned)(ga >> 32) | 0x80000000u;        // global_addr[56:32] | type=2
    uint32x8 g1;
    g1.s0 = 3u << 16;            // workgroup_mask=0, data_size=3 (8B)
    g1.s1 = 0x08000000u;         // tensor_dim0 = 2048  (bits 79:48)
    g1.s2 = 0x00010000u;         // tensor_dim1 = 1     (bits 111:80)
    g1.s3 = 0x08000000u;         // tile_dim0 = 2048    (bits 127:112)
    g1.s4 = 1u;                  // tile_dim1 = 1, tile_dim2 = 0
    g1.s5 = 2048u;               // tensor_dim0_stride = 2048 (bits 207:160)
    g1.s6 = 0u;
    g1.s7 = 0u;
    asm volatile("tensor_load_to_lds %0, %1" :: "s"(g0), "s"(g1) : "memory");
    __builtin_amdgcn_s_wait_tensorcnt(0);
  }
  __syncthreads();                                    // publish TDM tile
  {
    float4 v = soff[tid];
    sarea[tid] = (v.w - v.y + 1.0f) * (v.z - v.x + 1.0f);
  }
  __syncthreads();

  // (b) IoU bitmask: wave wv covers rows [wv*32, wv*32+32)
  for (int rr = 0; rr < 32; ++rr) {
    const int r = wv * 32 + rr;
    float4 a = soff[r]; float aa = sarea[r];
    unsigned myw = 0u;
    for (int cw = 0; cw < 32; ++cw) {
      int col = cw * 32 + lane;
      float4 c = soff[col]; float ca = sarea[col];
      float yy1 = fmaxf(a.x, c.x), xx1 = fmaxf(a.y, c.y);
      float yy2 = fminf(a.z, c.z), xx2 = fminf(a.w, c.w);
      float w = fmaxf(0.0f, xx2 - xx1 + 1.0f);
      float h = fmaxf(0.0f, yy2 - yy1 + 1.0f);
      float inter = w * h;
      bool pred = inter > IOU_TH * (aa + ca - inter);   // iou > 0.7, div-free
      unsigned bal = (unsigned)__ballot((int)pred);
      if (cw == lane) myw = bal;
    }
    sup[r * 32 + lane] = myw;
  }
  __syncthreads();

  // (c) greedy scan by wave 0: removed set = 1 word/lane, row prefetch,
  //     v_readlane (not ds_bpermute) on the serial critical chain.
  if (wv == 0) {
    unsigned removed = 0u, kw = 0u;
    unsigned nextrow = sup[lane];
    for (int i = 0; i < 1024; ++i) {
      unsigned row = nextrow;
      if (i < 1023) nextrow = sup[(i + 1) * 32 + lane];   // off critical path
      unsigned rwd = __builtin_amdgcn_readlane(removed, (unsigned)(i >> 5));
      if (!((rwd >> (i & 31)) & 1u)) {
        if (lane == (i >> 5)) kw |= 1u << (i & 31);
        removed |= row;
      }
    }
    keepw_s[lane] = kw;
  }
  __syncthreads();

  // (d) compaction + outputs
  bool kept = (keepw_s[wv] >> lane) & 1u;
  unsigned wm = (unsigned)__ballot((int)kept);
  if (lane == 0) wcnt[wv] = __popc(wm);
  __syncthreads();

  // WMMA prefix-scan of the 32 wave counts (wave 0, full EXEC as required).
  // D[m][n] = sum_k A[m][k]*B[k][n]; A[m][k] = wcnt[k]; B1[k][n] = [k<=n],
  // B2[k][n] = [k<=n+16]  =>  D[0][n] = inclusive prefix s_n / s_{n+16}.
  // Counts <= 32, sums <= 1024: exact in f16 inputs / f32 accumulation.
  // 16-bit operand K striping (ISA 7.12.2): lanes 0-15 hold K in {0..7,16..23},
  // lanes 16-31 hold K in {8..15,24..31}; lane mod 16 selects row m / col n.
  if (wv == 0) {
    const int klo = (lane < 16) ? 0 : 8;      // halves 0..7  -> K = klo + e
    const int khi = (lane < 16) ? 16 : 24;    // halves 8..15 -> K = khi + e
    const int col = lane & 15;
    v16h a, b1, b2;
    #pragma unroll
    for (int e = 0; e < 8; ++e) {
      int k0 = klo + e, k1 = khi + e;
      a[e]     = (_Float16)(float)wcnt[k0];
      a[8 + e] = (_Float16)(float)wcnt[k1];
      b1[e]     = (_Float16)((k0 <= col)      ? 1.0f : 0.0f);
      b1[8 + e] = (_Float16)((k1 <= col)      ? 1.0f : 0.0f);
      b2[e]     = (_Float16)((k0 <= col + 16) ? 1.0f : 0.0f);
      b2[8 + e] = (_Float16)((k1 <= col + 16) ? 1.0f : 0.0f);
    }
    v8f c1 = {}, c2 = {};
    c1 = __builtin_amdgcn_wmma_f32_16x16x32_f16(false, a, false, b1,
                                                (short)0, c1, false, false);
    c2 = __builtin_amdgcn_wmma_f32_16x16x32_f16(false, a, false, b2,
                                                (short)0, c2, false, false);
    if (lane < 16) {                 // D[0][n] lives in VGPR0 of lanes 0..15
      sincl[col]      = c1[0];
      sincl[col + 16] = c2[0];
    }
  }
  __syncthreads();

  const int nk   = (int)sincl[31];
  const int excl = (wv == 0) ? 0 : (int)sincl[wv - 1];
  const int keptBefore = excl + __popc(wm & ((1u << lane) - 1u));
  const int p = kept ? keptBefore : (nk + (tid - keptBefore));
  if (p < MAXDET) {
    float sc = kept ? wf[WS_TSCORE + b * 1024 + tid] : 0.0f;
    oscore[b * MAXDET + p] = sc;
    float4 bx = reinterpret_cast<const float4*>(wf + WS_TBOX)[b * 1024 + tid];
    reinterpret_cast<float4*>(obox)[b * MAXDET + p] = bx;
    olab[b * MAXDET + p] = ((int*)ws)[WS_TCLS + b * 1024 + tid];
  }
  if (tid == 0) onval[b] = nk < MAXDET ? nk : MAXDET;
}

// ---------------- launcher ----------------
extern "C" void kernel_launch(void* const* d_in, const int* in_sizes, int n_in,
                              void* d_out, int out_size, void* d_ws, size_t ws_size,
                              hipStream_t stream) {
  const float* boxes  = (const float*)d_in[0];   // (32, 8400, 4) f32
  const float* scores = (const float*)d_in[1];   // (32, 8400, 80) f32
  unsigned* ws = (unsigned*)d_ws;
  float* obox   = (float*)d_out;                             // (32,300,4)
  float* oscore = (float*)d_out + BATCH * MAXDET * 4;        // (32,300)
  int*   olab   = (int*)d_out + BATCH * MAXDET * 5;          // (32,300) i32
  int*   onval  = (int*)d_out + BATCH * MAXDET * 6;          // (32,1)   i32

  nms_init<<<dim3(256), 256, 0, stream>>>(ws);
  for (int r = 0; r < 2; ++r) {
    nms_hist<<<dim3(HBLK, BATCH), 256, 0, stream>>>(scores, ws, r);
    nms_scan_bins<<<dim3(BATCH), 32, 0, stream>>>(ws, r);
  }
  nms_collect<<<dim3(HBLK, BATCH), 256, 0, stream>>>(scores, ws);
  nms_sort<<<dim3(BATCH), 1024, 0, stream>>>(boxes, ws);
  nms_tail<<<dim3(BATCH), 1024, 0, stream>>>(ws, obox, oscore, olab, onval);
}
